// MoE_58256936403623
// MI455X (gfx1250) — matrix operand order repaired and checked by reading, not compiled, not run on hardware
//
#include <hip/hip_runtime.h>
#include <hip/hip_bf16.h>

// MoE top-2 (B=4,T=2048,C=512,H=2048,E=8,K=2) for MI455X / gfx1250.
// bf16 WMMA grouped-GEMM over gathered tokens per expert.
// d_out layout: logits[B*T*E] f32 | topk_idx[B*T*K] i32 | out[B*T*C] f32.
// Workspace (~42.5 MB): counters, tile offsets, per-expert token/gate lists,
// bf16 x copy, fragment-major bf16 W1/W2 copies.

#define Bb   4
#define Tt   2048
#define Cc   512
#define Hh   2048
#define Ee   8
#define Kk   2
#define NT   (Bb * Tt)          // 8192 tokens
#define CAP  NT                 // worst-case tokens per expert
#define MT   32                 // tokens per expert tile
#define PX   520                // x LDS pitch in halfwords (512 + pad)
#define PH   136                // h LDS pitch in halfwords (128 + pad)

typedef unsigned short ushort_t;
typedef unsigned int   uint_t;
typedef unsigned long long u64_t;

typedef __attribute__((ext_vector_type(16))) __bf16 v16bf;
typedef __attribute__((ext_vector_type(8)))  float  v8f;

union Frag { v16bf v; uint_t u[8]; uint4 q[2]; };

__device__ __forceinline__ ushort_t f2bf(float f) {
    uint_t u = __float_as_uint(f);
    u += 0x7FFFu + ((u >> 16) & 1u);          // round-to-nearest-even
    return (ushort_t)(u >> 16);
}
__device__ __forceinline__ uint_t pack2(ushort_t lo, ushort_t hi) {
    return (uint_t)lo | ((uint_t)hi << 16);
}

// ---------------------------------------------------------------- zero + cnt
__global__ void moe_zero(float* __restrict__ out, int n, int* __restrict__ cnt) {
    int i = blockIdx.x * blockDim.x + threadIdx.x;
    if (i < Ee) cnt[i] = 0;
    for (; i < n; i += gridDim.x * blockDim.x) out[i] = 0.0f;
}

// ---------------------------------------------------------------- conversions
__global__ void moe_cvt_x(const float* __restrict__ x, ushort_t* __restrict__ xb, int n) {
    for (int i = blockIdx.x * blockDim.x + threadIdx.x; i < n; i += gridDim.x * blockDim.x)
        xb[i] = f2bf(x[i]);
}

// W1[e][c][h] -> fragment-major W1f[e][ks=C/32][nt=H/16][lane=32][v=8] dwords.
// Lane (g=lane>>4, n=lane&15), dword v = (bf16 W1[k][h], bf16 W1[k+1][h]),
// k = ks*32 + g*8 + 2v + (v>=4 ? 8 : 0), h = nt*16 + n.  One fragment = 1 KB contiguous.
__global__ void moe_cvt_w1(const float* __restrict__ W1, uint_t* __restrict__ W1f, int n) {
    for (int i = blockIdx.x * blockDim.x + threadIdx.x; i < n; i += gridDim.x * blockDim.x) {
        int v    = i & 7;
        int lane = (i >> 3) & 31;
        int nt   = (i >> 8) & (Hh / 16 - 1);     // 128 n-tiles
        int ks   = (i >> 15) & (Cc / 32 - 1);    // 16 k-steps
        int e    = i >> 19;
        int g = lane >> 4, nn = lane & 15;
        int k = ks * 32 + g * 8 + 2 * v + (v >= 4 ? 8 : 0);
        int h = nt * 16 + nn;
        const float* p = W1 + ((size_t)e * Cc + k) * Hh + h;
        W1f[i] = pack2(f2bf(p[0]), f2bf(p[Hh]));
    }
}

// W2[e][h][c] -> fragment-major W2f[e][ks=H/32][nt=C/16][lane][v]
__global__ void moe_cvt_w2(const float* __restrict__ W2, uint_t* __restrict__ W2f, int n) {
    for (int i = blockIdx.x * blockDim.x + threadIdx.x; i < n; i += gridDim.x * blockDim.x) {
        int v    = i & 7;
        int lane = (i >> 3) & 31;
        int nt   = (i >> 8) & (Cc / 16 - 1);     // 32 n-tiles
        int ks   = (i >> 13) & (Hh / 32 - 1);    // 64 k-steps
        int e    = i >> 19;
        int g = lane >> 4, nn = lane & 15;
        int k = ks * 32 + g * 8 + 2 * v + (v >= 4 ? 8 : 0);
        int c = nt * 16 + nn;
        const float* p = W2 + ((size_t)e * Hh + k) * Cc + c;
        W2f[i] = pack2(f2bf(p[0]), f2bf(p[Cc]));
    }
}

// ---------------------------------------------------------------- router
__global__ void moe_router(const float* __restrict__ x, const float* __restrict__ Wr,
                           const float* __restrict__ br, float* __restrict__ logits_out,
                           int* __restrict__ idx_out, int* __restrict__ cnt,
                           int* __restrict__ tok_list, float* __restrict__ gate_list) {
    __shared__ float wr[Cc * Ee];                 // 16 KB
    for (int i = threadIdx.x; i < Cc * Ee; i += blockDim.x) wr[i] = Wr[i];
    __syncthreads();

    int t = blockIdx.x * blockDim.x + threadIdx.x;   // one token per thread
    float acc[Ee];
#pragma unroll
    for (int e = 0; e < Ee; ++e) acc[e] = br[e];
    const float* xr = x + (size_t)t * Cc;
    for (int c = 0; c < Cc; ++c) {
        float xv = xr[c];
#pragma unroll
        for (int e = 0; e < Ee; ++e) acc[e] = fmaf(xv, wr[c * Ee + e], acc[e]);
    }
#pragma unroll
    for (int e = 0; e < Ee; ++e) logits_out[(size_t)t * Ee + e] = acc[e];

    // top-2 (first max wins ties, matching jax.lax.top_k)
    int i1 = 0; float v1 = acc[0];
#pragma unroll
    for (int e = 1; e < Ee; ++e) if (acc[e] > v1) { v1 = acc[e]; i1 = e; }
    int i2 = (i1 == 0) ? 1 : 0; float v2 = acc[i2];
#pragma unroll
    for (int e = 0; e < Ee; ++e) if (e != i1 && acc[e] > v2) { v2 = acc[e]; i2 = e; }

    idx_out[t * Kk + 0] = i1;
    idx_out[t * Kk + 1] = i2;

    float s1 = 1.0f / (1.0f + __expf(v2 - v1));   // softmax over {v1,v2}
    float s2 = 1.0f - s1;

    int p1 = atomicAdd(&cnt[i1], 1);
    tok_list[i1 * CAP + p1]  = t;
    gate_list[i1 * CAP + p1] = s1;
    int p2 = atomicAdd(&cnt[i2], 1);
    tok_list[i2 * CAP + p2]  = t;
    gate_list[i2 * CAP + p2] = s2;
}

// ---------------------------------------------------------------- tile prefix
__global__ void moe_tileoff(const int* __restrict__ cnt, int* __restrict__ tileoff) {
    if (threadIdx.x == 0 && blockIdx.x == 0) {
        int s = 0;
        for (int e = 0; e < Ee; ++e) { tileoff[e] = s; s += (cnt[e] + MT - 1) / MT; }
        tileoff[Ee] = s;
    }
}

// ---------------------------------------------------------------- expert FFN
// 8 waves. Tile: 32 gathered tokens x 512 output cols, H chunked by 128.
// Wave w: token half mt=w&1, output col group ng=w>>1 (128 cols = 8 n-tiles).
__global__ __launch_bounds__(256) void moe_expert(
    const ushort_t* __restrict__ xb, const uint_t* __restrict__ W1f,
    const uint_t* __restrict__ W2f, const float* __restrict__ b1,
    const float* __restrict__ b2, const int* __restrict__ cnt,
    const int* __restrict__ tileoff, const int* __restrict__ tok_list,
    const float* __restrict__ gate_list, float* __restrict__ out) {

    __shared__ ushort_t xs[MT * PX];   // 33,280 B : gathered x tile (bf16)
    __shared__ ushort_t hs[MT * PH];   //  8,704 B : h chunk (bf16)
    __shared__ int      s_tok[MT];
    __shared__ float    s_gate[MT];

    int b = blockIdx.x;
    int total = tileoff[Ee];
    if (b >= total) return;
    int e = 0;
    while (b >= tileoff[e + 1]) ++e;
    int tile  = b - tileoff[e];
    int count = cnt[e];

    int tid = threadIdx.x;
    if (tid < MT) {
        int idx = tile * MT + tid;
        int tk = 0; float gg = 0.0f;
        if (idx < count) { tk = tok_list[e * CAP + idx]; gg = gate_list[e * CAP + idx]; }
        s_tok[tid] = tk; s_gate[tid] = gg;
    }
    __syncthreads();

    // gather 32 x-rows (1 KB each) into LDS with async global->LDS b128 copies
    // (ASYNCcnt-tracked, no VGPR round trip). 8 threads/row, 8 x 16B each.
    {
        int row = tid >> 3, q = tid & 7;
        u64_t  gbase = (u64_t)(const void*)(xb + (size_t)s_tok[row] * Cc);
        uint_t lbase = (uint_t)(uintptr_t)(const void*)(&xs[row * PX]);  // LDS = addr[31:0]
#pragma unroll
        for (int i = 0; i < 8; ++i) {
            uint_t off   = (uint_t)(q + i * 8) * 16u;
            uint_t laddr = lbase + off;
            u64_t  gaddr = gbase + off;
            asm volatile("global_load_async_to_lds_b128 %0, %1, off"
                         :: "v"(laddr), "v"(gaddr) : "memory");
        }
        asm volatile("s_wait_asynccnt 0x0" ::: "memory");
    }

    int wave = tid >> 5, lane = tid & 31;
    int g = lane >> 4, ln = lane & 15;
    int mt = wave & 1;          // token tile (rows mt*16 .. mt*16+15)
    int ng = wave >> 1;         // output col group: cols ng*128 .. +127

    // 16-bit A-fragment K offsets per VGPR (doc 7.12.2), plus g*8
    const int kA[8] = {0, 2, 4, 6, 16, 18, 20, 22};

    v8f acc[8];
#pragma unroll
    for (int j = 0; j < 8; ++j)
#pragma unroll
        for (int r = 0; r < 8; ++r) acc[j][r] = 0.0f;

    for (int hc = 0; hc < Hh; hc += 128) {
        __syncthreads();   // hs reusable; also publishes async-filled xs (iter 0)

        if (lane == 0)     // stream hint for this chunk's W2 panel
            __builtin_prefetch(W2f + (((size_t)e * (Hh / 32) + (hc >> 5)) *
                                      (Cc / 16)) * 256, 0, 1);

        // ---- GEMM1: h[32x128] = relu(x[32x512] @ W1[512x128] + b1)
        // wave computes tiles (mt, ht0) and (mt, ht1); A loaded once per k-step
        int nt0 = (hc >> 4) + (wave >> 1) * 2;   // global n-tile of first h-tile
        v8f c1a, c1b;
#pragma unroll
        for (int r = 0; r < 8; ++r) { c1a[r] = 0.0f; c1b[r] = 0.0f; }

        for (int ks = 0; ks < Cc; ks += 32) {
            Frag a, bf0, bf1;
            const ushort_t* ab = &xs[(mt * 16 + ln) * PX + ks + g * 8];
#pragma unroll
            for (int v = 0; v < 8; ++v) a.u[v] = *(const uint_t*)(ab + kA[v]);
            // fragment-major weights: one fragment = 32 lanes x 8 dwords
            const uint4* f0 = (const uint4*)(W1f +
                ((((size_t)e * (Cc / 32) + (ks >> 5)) * (Hh / 16) + nt0) * 32 + lane) * 8);
            const uint4* f1 = f0 + 64;           // next n-tile, +1 KB
            bf0.q[0] = f0[0]; bf0.q[1] = f0[1];
            bf1.q[0] = f1[0]; bf1.q[1] = f1[1];
            c1a = __builtin_amdgcn_wmma_f32_16x16x32_bf16(
                    false, a.v, false, bf0.v, (short)0, c1a, false, false);
            c1b = __builtin_amdgcn_wmma_f32_16x16x32_bf16(
                    false, a.v, false, bf1.v, (short)0, c1b, false, false);
        }
        // bias + relu + store h tiles to LDS as bf16
#pragma unroll
        for (int u = 0; u < 2; ++u) {
            int colg = (nt0 + u) * 16 + ln;      // global h column
            int colL = colg - hc;                // 0..127 within chunk
            float bias = b1[e * Hh + colg];
            v8f cu = u ? c1b : c1a;
#pragma unroll
            for (int r = 0; r < 8; ++r) {
                float hv = cu[r] + bias;
                hv = hv > 0.0f ? hv : 0.0f;      // relu
                int m = mt * 16 + g * 8 + r;     // D layout: M = r + g*8
                hs[m * PH + colL] = f2bf(hv);
            }
        }
        __syncthreads();

        // ---- GEMM2: acc += h[32x128] @ W2[128x512] (wave's 128 cols)
        for (int ks = 0; ks < 128; ks += 32) {
            Frag a;
            const ushort_t* ab = &hs[(mt * 16 + ln) * PH + ks + g * 8];
#pragma unroll
            for (int v = 0; v < 8; ++v) a.u[v] = *(const uint_t*)(ab + kA[v]);
            size_t kbase = ((size_t)e * (Hh / 32) + ((hc + ks) >> 5)) * (Cc / 16);
#pragma unroll
            for (int j = 0; j < 8; ++j) {
                Frag bf;
                const uint4* fp = (const uint4*)(W2f +
                    ((kbase + ng * 8 + j) * 32 + lane) * 8);
                bf.q[0] = fp[0]; bf.q[1] = fp[1];
                acc[j] = __builtin_amdgcn_wmma_f32_16x16x32_bf16(
                            false, a.v, false, bf.v, (short)0, acc[j], false, false);
            }
        }
    }

    // ---- epilogue: (acc + b2) * gate, scatter-accumulate into out
#pragma unroll
    for (int j = 0; j < 8; ++j) {
        int col = ng * 128 + j * 16 + ln;
        float bias = b2[e * Cc + col];
#pragma unroll
        for (int r = 0; r < 8; ++r) {
            int m = mt * 16 + g * 8 + r;
            float val = (acc[j][r] + bias) * s_gate[m];
            atomicAdd(&out[(size_t)s_tok[m] * Cc + col], val);
        }
    }
}

// ---------------------------------------------------------------- launch
extern "C" void kernel_launch(void* const* d_in, const int* in_sizes, int n_in,
                              void* d_out, int out_size, void* d_ws, size_t ws_size,
                              hipStream_t stream) {
    const float* x  = (const float*)d_in[0];
    const float* Wr = (const float*)d_in[1];
    const float* br = (const float*)d_in[2];
    const float* W1 = (const float*)d_in[3];
    const float* b1 = (const float*)d_in[4];
    const float* W2 = (const float*)d_in[5];
    const float* b2 = (const float*)d_in[6];

    float* logits = (float*)d_out;                       // [NT*Ee]
    int*   idx    = (int*)((char*)d_out + (size_t)NT * Ee * 4);        // [NT*Kk]
    float* out    = (float*)((char*)d_out + (size_t)(NT * Ee + NT * Kk) * 4); // [NT*Cc]

    char* w = (char*)d_ws;
    int*      cnt       = (int*)w;        w += 256;
    int*      tileoff   = (int*)w;        w += 256;
    int*      tok_list  = (int*)w;        w += (size_t)Ee * CAP * 4;   // 256 KB
    float*    gate_list = (float*)w;      w += (size_t)Ee * CAP * 4;   // 256 KB
    ushort_t* xb        = (ushort_t*)w;   w += (size_t)NT * Cc * 2;    // 8.4 MB
    uint_t*   W1f       = (uint_t*)w;     w += (size_t)Ee * (Cc / 2) * Hh * 4; // 16.8 MB
    uint_t*   W2f       = (uint_t*)w;     /* += 16.8 MB; total ~42.5 MB */

    const int nOut = NT * Cc;
    moe_zero<<<2048, 256, 0, stream>>>(out, nOut, cnt);

    const int nX = NT * Cc;
    moe_cvt_x<<<2048, 256, 0, stream>>>(x, xb, nX);
    const int nW1 = Ee * (Cc / 2) * Hh;      // dwords in W1f
    moe_cvt_w1<<<2048, 256, 0, stream>>>(W1, W1f, nW1);
    const int nW2 = Ee * (Hh / 2) * Cc;      // dwords in W2f
    moe_cvt_w2<<<2048, 256, 0, stream>>>(W2, W2f, nW2);

    moe_router<<<NT / 256, 256, 0, stream>>>(x, Wr, br, logits, idx,
                                             cnt, tok_list, gate_list);
    moe_tileoff<<<1, 1, 0, stream>>>(cnt, tileoff);

    // max tiles = sum_e ceil(cnt_e/32) <= 16384/32 + 8 = 520
    moe_expert<<<520, 256, 0, stream>>>(xb, W1f, W2f, b1, b2, cnt, tileoff,
                                        tok_list, gate_list, out);
}
